// PredictorModel_89111981458016
// MI455X (gfx1250) — compile-verified, matrix operand order checked by reading, not run on hardware
//
#include <hip/hip_runtime.h>
#include <math.h>
#include <stdint.h>

typedef _Float16 f16;
typedef _Float16 v16h __attribute__((ext_vector_type(16)));
typedef _Float16 v8h  __attribute__((ext_vector_type(8)));
typedef float    v8f  __attribute__((ext_vector_type(8)));

constexpr int HH = 144;   // image height
constexpr int WW = 256;   // image width (power of two; 64 | WW so a 64-row M-tile
                          // stays within one image row => single y per A-tile row)

#define WMMA_F32_F16(a, b, c) \
  __builtin_amdgcn_wmma_f32_16x16x32_f16(false, (a), false, (b), (short)0, (c), false, false)

#if __has_builtin(__builtin_amdgcn_tensor_load_to_lds)
#define HAVE_TDM 1
typedef unsigned int u32x4 __attribute__((ext_vector_type(4)));
typedef int i32x4 __attribute__((ext_vector_type(4)));
typedef int i32x8 __attribute__((ext_vector_type(8)));
#else
#define HAVE_TDM 0
#endif

__device__ __forceinline__ float hard_sigmoid(float x) {
  return fminf(fmaxf(0.2f * x + 0.5f, 0.0f), 1.0f);
}

// ---------------------------------------------------------------------------
// Elementwise utility kernels
// ---------------------------------------------------------------------------
__global__ void f32_to_f16_kernel(const float* __restrict__ in, f16* __restrict__ out, long n) {
  long i = (long)blockIdx.x * blockDim.x + threadIdx.x;
  if (i < n) out[i] = (f16)in[i];
}

__global__ void fill_f16_kernel(f16* __restrict__ p, long n) {
  long i = (long)blockIdx.x * blockDim.x + threadIdx.x;
  if (i < n) p[i] = (f16)0.0f;
}

__global__ void fill_f32_kernel(float* __restrict__ p, long n) {
  long i = (long)blockIdx.x * blockDim.x + threadIdx.x;
  if (i < n) p[i] = 0.0f;
}

// HWIO weights [K = kh*kw*Cin][N = Cout] (N fastest) -> transposed fp16 [Npad][Kpad]
__global__ void wtrans_kernel(const float* __restrict__ w, f16* __restrict__ out,
                              int K, int Kpad, int N, int Npad) {
  long i = (long)blockIdx.x * blockDim.x + threadIdx.x;
  long total = (long)Npad * Kpad;
  if (i >= total) return;
  int n = (int)(i / Kpad);
  int k = (int)(i % Kpad);
  float v = (k < K && n < N) ? w[(long)k * N + n] : 0.0f;
  out[i] = (f16)v;
}

// BatchNorm inference (eps=1e-3), h layout [M][64] fp16 -> fp16
__global__ void bn_kernel(const f16* __restrict__ h, const float* __restrict__ g,
                          const float* __restrict__ b, const float* __restrict__ mean,
                          const float* __restrict__ var, f16* __restrict__ out, long n) {
  long i = (long)blockIdx.x * blockDim.x + threadIdx.x;
  if (i >= n) return;
  int f = (int)(i & 63);
  float v = (float)h[i];
  out[i] = (f16)((v - mean[f]) * rsqrtf(var[f] + 1e-3f) * g[f] + b[f]);
}

// ---------------------------------------------------------------------------
// Implicit-GEMM tile staging
// ---------------------------------------------------------------------------
// FAST path (CIN % 32 == 0, KPAD == KREAL): a 32-wide K chunk lives at a single
// patch offset (ky,kx) and is contiguous in memory -> one global_load_b128 +
// one ds_store_b128 per thread for the whole 64x32 tile.
template <int CIN, int KW, int PAD>
__device__ __forceinline__ void load_a_tile_fast(const f16* __restrict__ src, long bstride,
                                                 f16* atile, int m0, int k0) {
  static_assert(CIN % 32 == 0, "fast A path needs CIN %% 32 == 0");
  const int kpatch = k0 / CIN;          // uniform, constant divide
  const int ci0 = k0 % CIN;
  const int kx = kpatch % KW - PAD;
  const int ky = kpatch / KW - PAD;
  const int tid = threadIdx.x;
  const int r = tid >> 2;               // A-tile row 0..63
  const int c4 = tid & 3;               // 16-byte chunk within the 64-byte row
  const int m = m0 + r;
  const int x = (m & (WW - 1)) + kx;
  const int my = m >> 8;                // WW == 256
  const int y = my % HH + ky;
  const int b = my / HH;
  v8h v = {};
  if ((unsigned)y < (unsigned)HH && (unsigned)x < (unsigned)WW) {
    const f16* p = src + (long)b * bstride + ((long)y * WW + x) * CIN + ci0 + c4 * 8;
    v = *(const v8h*)p;
  }
  *(v8h*)(atile + r * 32 + c4 * 8) = v;
}

// Generic path (small CIN / padded K): scalar gather with compile-time geometry.
template <int CIN, int KW, int PAD, int KREAL>
__device__ __forceinline__ void load_a_tile_gen(const f16* __restrict__ src, long bstride,
                                                f16* atile, int m0, int k0) {
  for (int idx = threadIdx.x; idx < 64 * 32; idx += 256) {
    int r = idx >> 5;
    int k = k0 + (idx & 31);
    f16 v = (f16)0.0f;
    if (k < KREAL) {
      int m = m0 + r;
      int x = m & (WW - 1);
      int my = m >> 8;
      int y = my % HH;
      int b = my / HH;
      int ci = k % CIN;
      int kxy = k / CIN;
      int kx = kxy % KW;
      int ky = kxy / KW;
      int yy = y + ky - PAD;
      int xx = x + kx - PAD;
      if ((unsigned)yy < (unsigned)HH && (unsigned)xx < (unsigned)WW)
        v = src[(long)b * bstride + ((long)yy * WW + xx) * CIN + ci];
    }
    atile[idx] = v;
  }
}

// B tile fallback: NTILE rows x 32 K from [N][KPAD], 128-bit vector loads.
template <int KPAD, int NTILE>
__device__ __forceinline__ void load_b_tile(const f16* __restrict__ wT, f16* btile, int k0) {
  constexpr int chunks = NTILE * 4;
  for (int idx = threadIdx.x; idx < chunks; idx += 256) {
    int n = idx >> 2;
    int c4 = idx & 3;
    const f16* p = wT + (long)n * KPAD + k0 + c4 * 8;
    *(v8h*)(btile + n * 32 + c4 * 8) = *(const v8h*)p;
    if (k0 + 32 < KPAD) __builtin_prefetch(p + 32, 0, 1);  // global_prefetch_b8
  }
}

#if HAVE_TDM
// Tensor Data Mover: DMA a 2D weight tile (NTILE rows x 64 bytes, row stride
// KPAD*2 bytes) from global [N][KPAD] into contiguous LDS.  D# bitfields per
// CDNA5 ISA 8.3/8.4 (group0: count/lds_addr/global_addr/type=2; group1:
// data_size=2B, tensor_dim0=KPAD, tensor_dim1=NTILE, tile_dim0=32,
// tile_dim1=NTILE, tensor_dim0_stride=KPAD).
template <int KPAD, int NTILE>
__device__ __forceinline__ void tdm_load_b_tile(const f16* __restrict__ wT, f16* btile, int k0) {
  unsigned long long ga = (unsigned long long)(uintptr_t)(wT + k0);
  unsigned lds = (unsigned)(uintptr_t)btile;  // LDS byte offset = addr[31:0]
  u32x4 g0 = {1u,                                   // count=1, no gather
              lds,                                  // lds_addr
              (unsigned)(ga & 0xFFFFFFFFu),         // global_addr[31:0]
              (unsigned)(((ga >> 32) & 0x01FFFFFFu) | (2u << 30))};  // [56:32] | type=2
  i32x8 g1 = {(int)(1u << 16),                      // wg_mask=0, data_size=1 (2 bytes)
              (int)(((unsigned)KPAD & 0xFFFFu) << 16),      // tensor_dim0[15:0]
              (int)((((unsigned)KPAD >> 16) & 0xFFFFu) | ((unsigned)NTILE << 16)),
              (int)(32u << 16),                     // tile_dim0 = 32 elements
              (int)NTILE,                           // tile_dim1
              (int)KPAD,                            // tensor_dim0_stride[31:0]
              0, 0};
  i32x4 gz = {};
  i32x8 gz8 = {};
  __builtin_amdgcn_tensor_load_to_lds(g0, g1, gz, gz, gz8, 0);
}
#endif

// Stage a B tile: TDM (wave 0 issues, EXEC-independent) or vector-load fallback.
template <int KPAD, int NTILE>
__device__ __forceinline__ void stage_b_tile(const f16* __restrict__ wT, f16* btile, int k0,
                                             int wave) {
#if HAVE_TDM
  if (wave == 0) tdm_load_b_tile<KPAD, NTILE>(wT, btile, k0);
#else
  (void)wave;
  load_b_tile<KPAD, NTILE>(wT, btile, k0);
#endif
}

// Wait for outstanding TDM transfers (issuing wave only; no-op without TDM).
__device__ __forceinline__ void tdm_drain(int wave) {
#if HAVE_TDM
  if (wave == 0) __builtin_amdgcn_s_wait_tensorcnt(0);
#else
  (void)wave;
#endif
}

// 16x32 fp16 fragment per ISA layout:
// lane<16: row=lane,   K = 0..7 and 16..23 ; lane>=16: row=lane-16, K = 8..15 and 24..31
__device__ __forceinline__ v16h load_frag(const f16* tile, int row16base, int lane) {
  int r = row16base + (lane & 15);
  int koff = (lane < 16) ? 0 : 8;
  const f16* p = tile + r * 32 + koff;
  v8h lo = *(const v8h*)p;
  v8h hi = *(const v8h*)(p + 16);
  v16h out;
#pragma unroll
  for (int i = 0; i < 8; i++) {
    out[i] = lo[i];
    out[i + 8] = hi[i];
  }
  return out;
}

// Double-buffered K-loop for the 64x256 ConvLSTM gate tile.  Each wave owns
// 2 row-groups x 4 col-groups: 2 A-frags + 4 B-frags feed 8 WMMAs (max reuse).
template <int CIN, int KW, int PAD, int KREAL, int KPAD, bool FASTA>
__device__ __forceinline__ void gemm_phase_256(const f16* __restrict__ src, long bstride,
                                               const f16* __restrict__ wT, f16* a0, f16* a1,
                                               f16* b0, f16* b1, int m0, int rgb16, int cqb,
                                               int lane, int wave, v8f (&acc)[8]) {
  f16* at[2] = {a0, a1};
  f16* bt[2] = {b0, b1};
  __syncthreads();  // previous consumers of these LDS buffers are done
  if constexpr (FASTA)
    load_a_tile_fast<CIN, KW, PAD>(src, bstride, at[0], m0, 0);
  else
    load_a_tile_gen<CIN, KW, PAD, KREAL>(src, bstride, at[0], m0, 0);
  stage_b_tile<KPAD, 256>(wT, bt[0], 0, wave);
  int p = 0;
  for (int k0 = 0; k0 < KPAD; k0 += 32) {
    tdm_drain(wave);   // TDM into tiles[p] complete before publishing
    __syncthreads();   // tiles[p] ready; tiles[p^1] free
    if (k0 + 32 < KPAD) {
      if constexpr (FASTA)
        load_a_tile_fast<CIN, KW, PAD>(src, bstride, at[p ^ 1], m0, k0 + 32);
      else
        load_a_tile_gen<CIN, KW, PAD, KREAL>(src, bstride, at[p ^ 1], m0, k0 + 32);
      stage_b_tile<KPAD, 256>(wT, bt[p ^ 1], k0 + 32, wave);
    }
    v16h af0 = load_frag(at[p], rgb16, lane);
    v16h af1 = load_frag(at[p], rgb16 + 16, lane);
#pragma unroll
    for (int cj = 0; cj < 4; cj++) {
      v16h b = load_frag(bt[p], (cqb + cj) * 16, lane);
      acc[cj] = WMMA_F32_F16(af0, b, acc[cj]);
      acc[4 + cj] = WMMA_F32_F16(af1, b, acc[4 + cj]);
    }
    p ^= 1;
  }
}

// ---------------------------------------------------------------------------
// Fused ConvLSTM step: z = conv(x_t,Wx) + conv(h,Wh) (+b in epilogue), then
// gate math.  Block tile: M=64 positions x N=256 gate channels (all 4 gates
// kept on-chip -> no 151 MB z round-trip to HBM).
// ---------------------------------------------------------------------------
__global__ __launch_bounds__(256) void convlstm_step_kernel(
    const f16* __restrict__ x16, const f16* __restrict__ hprev,
    const float* __restrict__ cprev, const f16* __restrict__ WxT,
    const f16* __restrict__ WhT, const float* __restrict__ bias,
    f16* __restrict__ hnext, float* __restrict__ cnext, int t) {
  __shared__ __align__(16) unsigned char smem[64 * 256 * 4];  // 64 KB of 320 KB WGP LDS
  f16* a0 = (f16*)smem;                    // 64x32 fp16  (4 KB)
  f16* a1 = (f16*)(smem + 4096);           // 64x32 fp16  (4 KB)
  f16* b0 = (f16*)(smem + 8192);           // 256x32 fp16 (16 KB)
  f16* b1 = (f16*)(smem + 24576);          // 256x32 fp16 (16 KB)
  float* zbuf = (float*)smem;              // 64x256 f32  (64 KB, aliases tiles)

  const int lane = threadIdx.x & 31;
  const int wave = threadIdx.x >> 5;
  const int rgb16 = (wave & 1) * 32;       // first of 2 owned row groups (x16 rows)
  const int cqb = (wave >> 1) * 4;         // first of 4 owned column groups
  const int m0 = blockIdx.x * 64;

  v8f acc[8];
#pragma unroll
  for (int j = 0; j < 8; j++) {
    v8f z = {};
    acc[j] = z;
  }

  // phase 1: input conv 5x5x3 (K=75 pad 96) -- generic gather, 3 chunks
  const f16* xbase = x16 + (long)t * HH * WW * 3;
  const long xbstride = (long)10 * HH * WW * 3;  // T = 10
  gemm_phase_256<3, 5, 2, 75, 96, false>(xbase, xbstride, WxT, a0, a1, b0, b1, m0, rgb16, cqb,
                                         lane, wave, acc);

  // phase 2: recurrent conv 5x5x64 (K=1600) -- fast contiguous path, 50 chunks
  const long hbstride = (long)HH * WW * 64;
  gemm_phase_256<64, 5, 2, 1600, 1600, true>(hprev, hbstride, WhT, a0, a1, b0, b1, m0, rgb16,
                                             cqb, lane, wave, acc);

  __syncthreads();
  // spill gate pre-activations to LDS (C/D layout: m = vgpr + (lane>=16?8:0), n = lane&15)
  {
    int n = (lane & 15);
    int mb = (lane >= 16) ? 8 : 0;
#pragma unroll
    for (int ri = 0; ri < 2; ri++) {
      int mrow = rgb16 + ri * 16 + mb;
#pragma unroll
      for (int cj = 0; cj < 4; cj++) {
        int nn = (cqb + cj) * 16 + n;
#pragma unroll
        for (int i = 0; i < 8; i++) zbuf[(mrow + i) * 256 + nn] = acc[ri * 4 + cj][i];
      }
    }
  }
  __syncthreads();

  // LSTM pointwise (gate order i, f, c, o)
  for (int idx = threadIdx.x; idx < 64 * 64; idx += 256) {
    int r = idx >> 6;
    int f = idx & 63;
    float zi = zbuf[r * 256 + f] + bias[f];
    float zf = zbuf[r * 256 + 64 + f] + bias[64 + f];
    float zc = zbuf[r * 256 + 128 + f] + bias[128 + f];
    float zo = zbuf[r * 256 + 192 + f] + bias[192 + f];
    float ig = hard_sigmoid(zi);
    float fg = hard_sigmoid(zf);
    float og = hard_sigmoid(zo);
    long gm = (long)(m0 + r) * 64 + f;
    float cn = fg * cprev[gm] + ig * tanhf(zc);
    cnext[gm] = cn;
    hnext[gm] = (f16)(og * tanhf(cn));
  }
}

// ---------------------------------------------------------------------------
// Generic SAME-conv + bias + activation (implicit GEMM, WMMA, double-buffered).
// Block tile M=64 x N=NPAD (<=64).  ACT 0: relu->fp16, 1: sigmoid->f32.
// Wave w owns (rg = w&3, cg = w>>2) and (same rg, cg+2): shared A fragment.
// ---------------------------------------------------------------------------
template <int CIN, int KW, int PAD, int KREAL, int KPAD, int NPAD, int COUT, int ACT, bool FASTA>
__global__ __launch_bounds__(256) void conv2d_wmma_kernel(const f16* __restrict__ src,
                                                          const f16* __restrict__ wT,
                                                          const float* __restrict__ bias,
                                                          f16* __restrict__ out16,
                                                          float* __restrict__ outf) {
  constexpr int BBYTES = NPAD * 32 * 2;
  __shared__ __align__(16) unsigned char smem[8192 + 2 * BBYTES];
  f16* a0 = (f16*)smem;                    // 64x32 fp16 (4 KB)
  f16* a1 = (f16*)(smem + 4096);           // 64x32 fp16 (4 KB)
  f16* b0 = (f16*)(smem + 8192);           // NPADx32 fp16
  f16* b1 = (f16*)(smem + 8192 + BBYTES);  // NPADx32 fp16

  const int lane = threadIdx.x & 31;
  const int wave = threadIdx.x >> 5;
  const int m0 = blockIdx.x * 64;
  constexpr int NSUB = 4 * (NPAD >> 4);
  constexpr long BSTRIDE = (long)HH * WW * CIN;

  v8f acc[2];
  {
    v8f z = {};
    acc[0] = z;
    acc[1] = z;
  }
  const int rg = wave & 3;
  const int cg0 = wave >> 2;
  const bool own0 = wave < NSUB;
  const bool own1 = wave + 8 < NSUB;

  f16* at[2] = {a0, a1};
  f16* bt[2] = {b0, b1};
  if constexpr (FASTA)
    load_a_tile_fast<CIN, KW, PAD>(src, BSTRIDE, at[0], m0, 0);
  else
    load_a_tile_gen<CIN, KW, PAD, KREAL>(src, BSTRIDE, at[0], m0, 0);
  stage_b_tile<KPAD, NPAD>(wT, bt[0], 0, wave);
  int p = 0;
  for (int k0 = 0; k0 < KPAD; k0 += 32) {
    tdm_drain(wave);
    __syncthreads();
    if (k0 + 32 < KPAD) {
      if constexpr (FASTA)
        load_a_tile_fast<CIN, KW, PAD>(src, BSTRIDE, at[p ^ 1], m0, k0 + 32);
      else
        load_a_tile_gen<CIN, KW, PAD, KREAL>(src, BSTRIDE, at[p ^ 1], m0, k0 + 32);
      stage_b_tile<KPAD, NPAD>(wT, bt[p ^ 1], k0 + 32, wave);
    }
    if (own0) {
      v16h a = load_frag(at[p], rg * 16, lane);  // shared by both subtiles
      v16h bA = load_frag(bt[p], cg0 * 16, lane);
      acc[0] = WMMA_F32_F16(a, bA, acc[0]);
      if (own1) {
        v16h bB = load_frag(bt[p], (cg0 + 2) * 16, lane);
        acc[1] = WMMA_F32_F16(a, bB, acc[1]);
      }
    }
    p ^= 1;
  }

#pragma unroll
  for (int s = 0; s < 2; s++) {
    if (s == 0 ? !own0 : !own1) continue;
    int cg = cg0 + s * 2;
    int n = cg * 16 + (lane & 15);
    if (n >= COUT) continue;
    int mb = m0 + rg * 16 + ((lane >= 16) ? 8 : 0);
    float bv = bias[n];
#pragma unroll
    for (int i = 0; i < 8; i++) {
      float v = acc[s][i] + bv;
      long o = (long)(mb + i) * COUT + n;
      if (ACT == 0)
        out16[o] = (f16)fmaxf(v, 0.0f);
      else
        outf[o] = 1.0f / (1.0f + expf(-v));
    }
  }
}

// ---------------------------------------------------------------------------
// Host orchestration
// ---------------------------------------------------------------------------
extern "C" void kernel_launch(void* const* d_in, const int* in_sizes, int n_in,
                              void* d_out, int out_size, void* d_ws, size_t ws_size,
                              hipStream_t stream) {
  (void)in_sizes; (void)n_in; (void)out_size; (void)ws_size;
  const float* inputs = (const float*)d_in[0];
  const float* Wx = (const float*)d_in[1];
  const float* Wh = (const float*)d_in[2];
  const float* b_lstm = (const float*)d_in[3];
  const float* gamma = (const float*)d_in[4];
  const float* beta = (const float*)d_in[5];
  const float* mmean = (const float*)d_in[6];
  const float* mvar = (const float*)d_in[7];
  const float* W1 = (const float*)d_in[8];  const float* b1 = (const float*)d_in[9];
  const float* W2 = (const float*)d_in[10]; const float* b2 = (const float*)d_in[11];
  const float* W3 = (const float*)d_in[12]; const float* b3 = (const float*)d_in[13];
  const float* W4 = (const float*)d_in[14]; const float* b4 = (const float*)d_in[15];
  const float* W5 = (const float*)d_in[16]; const float* b5 = (const float*)d_in[17];
  float* outp = (float*)d_out;

  const int B = 4, T = 10, F = 64;
  const long M = (long)B * HH * WW;  // 147456 spatial positions

  unsigned char* ws = (unsigned char*)d_ws;
  size_t off = 0;
  auto alloc = [&](size_t bytes) -> void* {
    void* p = ws + off;
    off = (off + bytes + 255) & ~(size_t)255;
    return p;
  };

  f16* x16 = (f16*)alloc((size_t)B * T * HH * WW * 3 * sizeof(f16));
  f16* WxT = (f16*)alloc((size_t)256 * 96 * sizeof(f16));
  f16* WhT = (f16*)alloc((size_t)256 * 1600 * sizeof(f16));
  f16* W1T = (f16*)alloc((size_t)16 * 5184 * sizeof(f16));
  f16* W2T = (f16*)alloc((size_t)32 * 416 * sizeof(f16));
  f16* W3T = (f16*)alloc((size_t)64 * 288 * sizeof(f16));
  f16* W4T = (f16*)alloc((size_t)64 * 576 * sizeof(f16));
  f16* W5T = (f16*)alloc((size_t)16 * 576 * sizeof(f16));
  f16* hA = (f16*)alloc((size_t)M * F * sizeof(f16));
  f16* hB = (f16*)alloc((size_t)M * F * sizeof(f16));
  float* cbuf = (float*)alloc((size_t)M * F * sizeof(float));
  f16* bn16 = (f16*)alloc((size_t)M * F * sizeof(f16));
  f16* actA = (f16*)alloc((size_t)M * 64 * sizeof(f16));
  f16* actB = (f16*)alloc((size_t)M * 64 * sizeof(f16));

  // 1) Convert input sequence to fp16
  long nx = (long)B * T * HH * WW * 3;
  f32_to_f16_kernel<<<(unsigned)((nx + 255) / 256), 256, 0, stream>>>(inputs, x16, nx);

  // 2) Pre-transpose all conv weights to [N][Kpad] fp16 (zero-padded)
  auto wt = [&](const float* w, f16* o, int K, int Kpad, int N, int Npad) {
    long tot = (long)Npad * Kpad;
    wtrans_kernel<<<(unsigned)((tot + 255) / 256), 256, 0, stream>>>(w, o, K, Kpad, N, Npad);
  };
  wt(Wx, WxT, 75, 96, 256, 256);
  wt(Wh, WhT, 1600, 1600, 256, 256);
  wt(W1, W1T, 5184, 5184, 16, 16);
  wt(W2, W2T, 400, 416, 32, 32);
  wt(W3, W3T, 288, 288, 64, 64);
  wt(W4, W4T, 576, 576, 64, 64);
  wt(W5, W5T, 576, 576, 3, 16);

  // 3) Zero-init LSTM state
  long nh = M * F;
  fill_f16_kernel<<<(unsigned)((nh + 255) / 256), 256, 0, stream>>>(hA, nh);
  fill_f32_kernel<<<(unsigned)((nh + 255) / 256), 256, 0, stream>>>(cbuf, nh);

  // 4) ConvLSTM recurrence (h ping-pong fp16, c in place f32)
  const int mblocks = (int)(M / 64);  // 2304
  f16* hprev = hA;
  f16* hnext = hB;
  for (int t = 0; t < T; t++) {
    convlstm_step_kernel<<<mblocks, 256, 0, stream>>>(x16, hprev, cbuf, WxT, WhT, b_lstm,
                                                      hnext, cbuf, t);
    f16* tmp = hprev;
    hprev = hnext;
    hnext = tmp;
  }

  // 5) BatchNorm (inference)
  bn_kernel<<<(unsigned)((nh + 255) / 256), 256, 0, stream>>>(hprev, gamma, beta, mmean, mvar,
                                                              bn16, nh);

  // 6) Conv stack (WMMA implicit GEMM)  <CIN,KW,PAD,KREAL,KPAD,NPAD,COUT,ACT,FASTA>
  conv2d_wmma_kernel<64, 9, 4, 5184, 5184, 16, 16, 0, true>
      <<<mblocks, 256, 0, stream>>>(bn16, W1T, b1, actA, nullptr);
  conv2d_wmma_kernel<16, 5, 2, 400, 416, 32, 32, 0, false>
      <<<mblocks, 256, 0, stream>>>(actA, W2T, b2, actB, nullptr);
  conv2d_wmma_kernel<32, 3, 1, 288, 288, 64, 64, 0, true>
      <<<mblocks, 256, 0, stream>>>(actB, W3T, b3, actA, nullptr);
  conv2d_wmma_kernel<64, 3, 1, 576, 576, 64, 64, 0, true>
      <<<mblocks, 256, 0, stream>>>(actA, W4T, b4, actB, nullptr);
  conv2d_wmma_kernel<64, 3, 1, 576, 576, 16, 3, 1, true>
      <<<mblocks, 256, 0, stream>>>(actB, W5T, b5, nullptr, outp);
}
